// DeltaNet_58205396795472
// MI455X (gfx1250) — matrix operand order repaired and checked by reading, not compiled, not run on hardware
//
#include <hip/hip_runtime.h>

// ---------------------------------------------------------------------------
// Types (plain ext vectors: trivially-copyable, union-safe)
// ---------------------------------------------------------------------------
typedef __attribute__((ext_vector_type(16))) __bf16      v16bf;
typedef __attribute__((ext_vector_type(8)))  float       v8f;
typedef __attribute__((ext_vector_type(4)))  unsigned int u32x4;
typedef __attribute__((ext_vector_type(2)))  unsigned int u32x2;
typedef __attribute__((ext_vector_type(4)))  float       f32x4;

#define S_LEN   2048
#define NHEAD   4
#define HIDDEN  1024
#define DKD     256
#define CHK     64            // delta-rule chunk length
#define NCHK    (S_LEN / CHK)
#define DVSL    128           // dv slice per workgroup (dv columns are independent)

// ---------------------------------------------------------------------------
// Helpers
// ---------------------------------------------------------------------------
__device__ __forceinline__ unsigned short f2bf(float x) {
  unsigned int u = __float_as_uint(x);
  u += 0x7FFFu + ((u >> 16) & 1u);              // round-to-nearest-even
  return (unsigned short)(u >> 16);
}

union FragBF { v16bf v; u32x4 q[2]; unsigned short s[16]; };

// A-operand fragment (16x32 bf16), storage row-major [rows][ld], element (m,k):
// lane = m + 16*((k%16)>=8); elems 0..7 -> k0+hi*8+{0..7}, elems 8..15 -> +16
__device__ __forceinline__ v16bf load_frag_a(const unsigned short* base, int ld,
                                             int r0, int k0, int lane) {
  int m = lane & 15, hi = lane >> 4;
  const unsigned short* p = base + (size_t)(r0 + m) * ld + k0 + hi * 8;
  FragBF f;
  f.q[0] = *(const u32x4*)(p);
  f.q[1] = *(const u32x4*)(p + 16);
  return f.v;
}

// B-operand fragment (32x16 bf16), storage is B^T row-major [n][ld]:
// lane = n + 16*((k%16)>=8); elems 0..7 -> k0+hi*8+{0..7}, elems 8..15 -> +16
__device__ __forceinline__ v16bf load_frag_b(const unsigned short* base, int ld,
                                             int n0, int k0, int lane) {
  int n = lane & 15, hi = lane >> 4;
  const unsigned short* p = base + (size_t)(n0 + n) * ld + k0 + hi * 8;
  FragBF f;
  f.q[0] = *(const u32x4*)(p);
  f.q[1] = *(const u32x4*)(p + 16);
  return f.v;
}

__device__ __forceinline__ v8f wmma_bf16(v16bf a, v16bf b, v8f c) {
  return __builtin_amdgcn_wmma_f32_16x16x32_bf16(false, a, false, b,
                                                 (short)0, c, false, false);
}

__device__ __forceinline__ void store8_bf16(unsigned short* dst, v8f x) {
  union { u32x4 q; unsigned short s[8]; } u;
#pragma unroll
  for (int i = 0; i < 8; i++) u.s[i] = f2bf(x[i]);
  *(u32x4*)dst = u.q;
}

// ---------------------------------------------------------------------------
// f32 -> bf16 (elementwise, vec4)
// ---------------------------------------------------------------------------
__global__ __launch_bounds__(256) void f32_to_bf16_kernel(
    const float* __restrict__ x, unsigned short* __restrict__ y, int n) {
  int i = (blockIdx.x * 256 + threadIdx.x) * 4;
  if (i + 3 < n) {
    f32x4 v = *(const f32x4*)(x + i);
    union { u32x2 d; unsigned short s[4]; } u;
    u.s[0] = f2bf(v.x); u.s[1] = f2bf(v.y); u.s[2] = f2bf(v.z); u.s[3] = f2bf(v.w);
    *(u32x2*)(y + i) = u.d;
  } else {
    for (; i < n; ++i) y[i] = f2bf(x[i]);
  }
}

// W[k*N+n] (f32, [K][N]) -> Wt[n*K+k] (bf16, [N][K])
__global__ __launch_bounds__(256) void transpose_to_bf16_kernel(
    const float* __restrict__ W, unsigned short* __restrict__ Wt, int K, int N) {
  int idx = blockIdx.x * 256 + threadIdx.x;
  if (idx >= K * N) return;
  int n = idx / K, k = idx - n * K;
  Wt[idx] = f2bf(W[(size_t)k * N + n]);
}

// ---------------------------------------------------------------------------
// GEMM: C[M,N] f32 = A_bf16[M,K] @ Bt_bf16[N,K]^T, fragments straight from L2
// block = 8 waves; wave computes 32x32; block tile 128x64
// ---------------------------------------------------------------------------
__global__ __launch_bounds__(256) void gemm_bf16(
    const unsigned short* __restrict__ A, const unsigned short* __restrict__ Bt,
    float* __restrict__ C, int M, int N, int K) {
  const int lane = threadIdx.x & 31, wv = threadIdx.x >> 5;
  const int Mb = blockIdx.y * 128 + (wv & 3) * 32;
  const int Nb = blockIdx.x * 64 + (wv >> 2) * 32;
  v8f zero8 = {};
  v8f acc[2][2];
  acc[0][0] = zero8; acc[0][1] = zero8; acc[1][0] = zero8; acc[1][1] = zero8;
  for (int k0 = 0; k0 < K; k0 += 32) {
    v16bf a0 = load_frag_a(A, K, Mb, k0, lane);
    v16bf a1 = load_frag_a(A, K, Mb + 16, k0, lane);
    v16bf b0 = load_frag_b(Bt, K, Nb, k0, lane);
    v16bf b1 = load_frag_b(Bt, K, Nb + 16, k0, lane);
    acc[0][0] = wmma_bf16(a0, b0, acc[0][0]);
    acc[0][1] = wmma_bf16(a0, b1, acc[0][1]);
    acc[1][0] = wmma_bf16(a1, b0, acc[1][0]);
    acc[1][1] = wmma_bf16(a1, b1, acc[1][1]);
  }
  const int n = lane & 15, hi = lane >> 4;
#pragma unroll
  for (int ti = 0; ti < 2; ++ti)
#pragma unroll
    for (int tj = 0; tj < 2; ++tj)
#pragma unroll
      for (int r = 0; r < 8; ++r)
        C[(size_t)(Mb + ti * 16 + hi * 8 + r) * N + Nb + tj * 16 + n] = acc[ti][tj][r];
}

// ---------------------------------------------------------------------------
// beta = sigmoid(hs @ Wb) ; one wave per row, wave32 shfl reduction
// ---------------------------------------------------------------------------
__global__ __launch_bounds__(256) void beta_kernel(
    const float* __restrict__ hs, const float* __restrict__ Wb,
    float* __restrict__ beta) {
  int row = blockIdx.x * 8 + (threadIdx.x >> 5);
  int lane = threadIdx.x & 31;
  float acc[NHEAD] = {0.f, 0.f, 0.f, 0.f};
  const float* hr = hs + (size_t)row * HIDDEN;
  for (int c = lane; c < HIDDEN; c += 32) {
    float h = hr[c];
#pragma unroll
    for (int j = 0; j < NHEAD; ++j) acc[j] += h * Wb[c * NHEAD + j];
  }
#pragma unroll
  for (int m = 16; m; m >>= 1)
#pragma unroll
    for (int j = 0; j < NHEAD; ++j) acc[j] += __shfl_xor(acc[j], m, 32);
  if (lane == 0)
#pragma unroll
    for (int j = 0; j < NHEAD; ++j)
      beta[(size_t)row * NHEAD + j] = 1.f / (1.f + __expf(-acc[j]));
}

// ---------------------------------------------------------------------------
// fused causal conv(K=4) + SiLU (+ L2 norm over head dim for q,k), bf16 out
// one wave per (row, head, tensor); lane owns 8 channels
// ---------------------------------------------------------------------------
__global__ __launch_bounds__(256) void conv_silu_norm_kernel(
    const float* __restrict__ qf, const float* __restrict__ kf,
    const float* __restrict__ vf, const float* __restrict__ wq,
    const float* __restrict__ wk, const float* __restrict__ wvv,
    unsigned short* __restrict__ qo, unsigned short* __restrict__ ko,
    unsigned short* __restrict__ vo) {
  int id = blockIdx.x * 8 + (threadIdx.x >> 5);
  int lane = threadIdx.x & 31;
  int row = id / 12, rem = id - row * 12;
  int tsel = rem >> 2, h = rem & 3;
  int b = row >> 11, s = row & 2047;
  const float* x = (tsel == 0) ? qf : (tsel == 1) ? kf : vf;
  const float* w = (tsel == 0) ? wq : (tsel == 1) ? wk : wvv;
  unsigned short* y = (tsel == 0) ? qo : (tsel == 1) ? ko : vo;
  int ch0 = h * DKD + lane * 8;
  float out[8] = {0.f, 0.f, 0.f, 0.f, 0.f, 0.f, 0.f, 0.f};
#pragma unroll
  for (int d = 0; d < 4; ++d) {
    int sp = s - 3 + d;
    if (sp < 0) continue;
    const float* xr = x + (size_t)(b * S_LEN + sp) * HIDDEN + ch0;
    f32x4 v0 = ((const f32x4*)xr)[0];
    f32x4 v1 = ((const f32x4*)xr)[1];
    float xv[8] = {v0.x, v0.y, v0.z, v0.w, v1.x, v1.y, v1.z, v1.w};
#pragma unroll
    for (int c = 0; c < 8; ++c) out[c] += w[(ch0 + c) * 4 + d] * xv[c];
  }
#pragma unroll
  for (int c = 0; c < 8; ++c) out[c] *= 1.f / (1.f + __expf(-out[c]));  // SiLU
  if (tsel < 2) {                                                       // l2norm
    float ss = 0.f;
#pragma unroll
    for (int c = 0; c < 8; ++c) ss += out[c] * out[c];
#pragma unroll
    for (int m = 16; m; m >>= 1) ss += __shfl_xor(ss, m, 32);
    float sc = rsqrtf(ss + 1e-6f);
#pragma unroll
    for (int c = 0; c < 8; ++c) out[c] *= sc;
  }
  union { u32x4 q; unsigned short s8[8]; } u;
#pragma unroll
  for (int c = 0; c < 8; ++c) u.s8[c] = f2bf(out[c]);
  *(u32x4*)(y + (size_t)row * HIDDEN + ch0) = u.q;
}

// ---------------------------------------------------------------------------
// Chunked delta rule.  Grid = B*NH*2 workgroups (dv split x2), 16 waves each.
// State S[dk=256, dv=128] lives in f32 VGPR accumulators (wave wv owns dk rows
// [16wv,16wv+16)); bf16 shadow St[dv][dk] in LDS is the WMMA B-operand.
// Per chunk (C=64): G=K K^T -> T=(I-tril(diag(b)G,-1))^-1 (fwd subst, 1 wave)
//   T' = T diag(b);  Wn = -(T'@K);  U' = T'@V + Wn@S;  Aqk = tril(Q K^T,0)
//   O  = Q@S + Aqk@U';   S += K^T @ U'
// ---------------------------------------------------------------------------
__global__ __launch_bounds__(512) void delta_kernel(
    const unsigned short* __restrict__ qg, const unsigned short* __restrict__ kg,
    const unsigned short* __restrict__ vg, const float* __restrict__ betag,
    float* __restrict__ og) {
  extern __shared__ char smem[];
  unsigned short* sQ  = (unsigned short*)smem;        // [64][256]
  unsigned short* sK  = sQ + 64 * DKD;                // [64][256]
  unsigned short* sKt = sK + 64 * DKD;                // [256][64]  K^T
  unsigned short* sVt = sKt + DKD * CHK;              // [128][64]  V^T (slice)
  unsigned short* sSt = sVt + DVSL * CHK;             // [128][256] S^T shadow
  unsigned short* sTb = sSt + DVSL * DKD;             // [64][64]   T' then Aqk
  unsigned short* sWU = sTb + CHK * CHK;              // [64][256]Wn -> [128][64]U'^T
  float* sTw = (float*)(sWU + 64 * DKD);              // [64][66]
  float* sBt = sTw + 64 * 66;                         // [64]

  const int tid = threadIdx.x, lane = tid & 31, wv = tid >> 5;
  const int g = blockIdx.x & 1;
  const int bh = blockIdx.x >> 1;
  const int b = bh >> 2, h = bh & 3;
  const int chq = h * DKD;               // q/k channel base
  const int chv = h * DKD + g * DVSL;    // v/o channel base (dv slice)

  v8f zero8 = {};
  v8f Sacc[8];
#pragma unroll
  for (int i = 0; i < 8; ++i) Sacc[i] = zero8;
  for (int i = tid; i < DVSL * DKD / 8; i += 512) ((u32x4*)sSt)[i] = (u32x4){0, 0, 0, 0};
  __syncthreads();

  for (int t = 0; t < NCHK; ++t) {
    const int rowb = b * S_LEN + t * CHK;

    // ---- stage q,k (row-major), k^T, v^T, beta ----
    {
      int r = tid >> 3;
      int c = (tid & 7) * 32;
      const unsigned short* gq = qg + (size_t)(rowb + r) * HIDDEN + chq + c;
      const unsigned short* gk = kg + (size_t)(rowb + r) * HIDDEN + chq + c;
      union { u32x4 q; unsigned short s8[8]; } kq[4];
#pragma unroll
      for (int i = 0; i < 4; ++i) {
        u32x4 qv = ((const u32x4*)gq)[i];
        ((u32x4*)(sQ + (size_t)r * DKD + c))[i] = qv;
        kq[i].q = ((const u32x4*)gk)[i];
        ((u32x4*)(sK + (size_t)r * DKD + c))[i] = kq[i].q;
      }
#pragma unroll
      for (int i = 0; i < 4; ++i)
#pragma unroll
        for (int e = 0; e < 8; ++e)
          sKt[(size_t)(c + i * 8 + e) * CHK + r] = kq[i].s8[e];

      int dv0 = (tid & 7) * 16;
      const unsigned short* gv = vg + (size_t)(rowb + r) * HIDDEN + chv + dv0;
      union { u32x4 q; unsigned short s8[8]; } vq[2];
      vq[0].q = ((const u32x4*)gv)[0];
      vq[1].q = ((const u32x4*)gv)[1];
#pragma unroll
      for (int i = 0; i < 2; ++i)
#pragma unroll
        for (int e = 0; e < 8; ++e)
          sVt[(size_t)(dv0 + i * 8 + e) * CHK + r] = vq[i].s8[e];

      if (tid < CHK) sBt[tid] = betag[(size_t)(rowb + tid) * NHEAD + h];
    }
    __syncthreads();

    // ---- G = K K^T ; write L = strict-lower(diag(beta) G) into sTw ----
    {
      int ti = wv >> 2, tj = wv & 3;
      v8f acc = zero8;
#pragma unroll
      for (int kk = 0; kk < 8; ++kk) {
        v16bf a = load_frag_a(sK, DKD, ti * 16, kk * 32, lane);
        v16bf bb = load_frag_b(sK, DKD, tj * 16, kk * 32, lane);
        acc = wmma_bf16(a, bb, acc);
      }
      int n = lane & 15, hi = lane >> 4;
#pragma unroll
      for (int r = 0; r < 8; ++r) {
        int i = ti * 16 + hi * 8 + r, j = tj * 16 + n;
        sTw[i * 66 + j] = (i > j) ? sBt[i] * acc[r] : 0.0f;
      }
    }
    __syncthreads();

    // ---- forward substitution: M = (I-L)^{-1} - I, in place (wave 0) ----
    if (wv == 0) {
      for (int i = 2; i < CHK; ++i) {
        float a0 = sTw[i * 66 + lane];
        float a1 = sTw[i * 66 + lane + 32];
        for (int j = 1; j < i; ++j) {
          float lij = sTw[i * 66 + j];           // broadcast read
          a0 += lij * sTw[j * 66 + lane];        // M[j][k]==0 for k>=j
          a1 += lij * sTw[j * 66 + lane + 32];
        }
        if (lane < i) sTw[i * 66 + lane] = a0;
        if (lane + 32 < i) sTw[i * 66 + lane + 32] = a1;
      }
    }
    __syncthreads();

    // ---- T' = (I + M) diag(beta), bf16 ----
    for (int idx = tid; idx < CHK * CHK; idx += 512) {
      int i = idx >> 6, j = idx & 63;
      float tv = (i == j) ? 1.0f : ((i > j) ? sTw[i * 66 + j] : 0.0f);
      sTb[idx] = f2bf(tv * sBt[j]);
    }
    __syncthreads();

    // ---- Wn = -(T' @ K)  [64,256] ----
    {
      int mt = wv & 3, ntb = (wv >> 2) * 4;
      for (int q4 = 0; q4 < 4; ++q4) {
        int nt = ntb + q4;
        v8f acc = zero8;
#pragma unroll
        for (int kk = 0; kk < 2; ++kk) {
          v16bf a = load_frag_a(sTb, CHK, mt * 16, kk * 32, lane);
          v16bf bb = load_frag_b(sKt, CHK, nt * 16, kk * 32, lane);
          acc = wmma_bf16(a, bb, acc);
        }
        int n = lane & 15, hi = lane >> 4;
#pragma unroll
        for (int r = 0; r < 8; ++r)
          sWU[(size_t)(mt * 16 + hi * 8 + r) * DKD + nt * 16 + n] = f2bf(-acc[r]);
      }
    }
    __syncthreads();

    // ---- U' = T'@V + Wn@S   (2 tiles / wave, kept in regs) ----
    v8f Uacc[2];
    {
      int mt = wv & 3, nt0 = (wv >> 2) * 2;
#pragma unroll
      for (int c = 0; c < 2; ++c) {
        int nt = nt0 + c;
        v8f acc = zero8;
#pragma unroll
        for (int kk = 0; kk < 2; ++kk) {
          v16bf a = load_frag_a(sTb, CHK, mt * 16, kk * 32, lane);
          v16bf bb = load_frag_b(sVt, CHK, nt * 16, kk * 32, lane);
          acc = wmma_bf16(a, bb, acc);
        }
#pragma unroll
        for (int kk = 0; kk < 8; ++kk) {
          v16bf a = load_frag_a(sWU, DKD, mt * 16, kk * 32, lane);
          v16bf bb = load_frag_b(sSt, DKD, nt * 16, kk * 32, lane);
          acc = wmma_bf16(a, bb, acc);
        }
        Uacc[c] = acc;
      }
    }
    __syncthreads();  // everyone finished reading Wn from sWU
    {                 // store U'^T [dv][64] over sWU
      int mt = wv & 3, nt0 = (wv >> 2) * 2;
      int n = lane & 15, hi = lane >> 4;
#pragma unroll
      for (int c = 0; c < 2; ++c) {
        int dv = (nt0 + c) * 16 + n;
        store8_bf16(&sWU[(size_t)dv * CHK + mt * 16 + hi * 8], Uacc[c]);
      }
    }
    __syncthreads();

    // ---- Aqk = tril(Q K^T, incl diag), bf16, overwrite sTb ----
    {
      int ti = wv >> 2, tj = wv & 3;
      v8f acc = zero8;
#pragma unroll
      for (int kk = 0; kk < 8; ++kk) {
        v16bf a = load_frag_a(sQ, DKD, ti * 16, kk * 32, lane);
        v16bf bb = load_frag_b(sK, DKD, tj * 16, kk * 32, lane);
        acc = wmma_bf16(a, bb, acc);
      }
      int n = lane & 15, hi = lane >> 4;
#pragma unroll
      for (int r = 0; r < 8; ++r) {
        int i = ti * 16 + hi * 8 + r, j = tj * 16 + n;
        sTb[i * CHK + j] = (j <= i) ? f2bf(acc[r]) : (unsigned short)0;
      }
    }
    __syncthreads();

    // ---- O = Q@S + Aqk@U'  -> global ----
    {
      int mt = wv & 3, nt0 = (wv >> 2) * 2;
#pragma unroll
      for (int c = 0; c < 2; ++c) {
        int nt = nt0 + c;
        v8f acc = zero8;
#pragma unroll
        for (int kk = 0; kk < 8; ++kk) {
          v16bf a = load_frag_a(sQ, DKD, mt * 16, kk * 32, lane);
          v16bf bb = load_frag_b(sSt, DKD, nt * 16, kk * 32, lane);
          acc = wmma_bf16(a, bb, acc);
        }
#pragma unroll
        for (int kk = 0; kk < 2; ++kk) {
          v16bf a = load_frag_a(sTb, CHK, mt * 16, kk * 32, lane);
          v16bf bb = load_frag_b(sWU, CHK, nt * 16, kk * 32, lane);
          acc = wmma_bf16(a, bb, acc);
        }
        int n = lane & 15, hi = lane >> 4;
#pragma unroll
        for (int r = 0; r < 8; ++r)
          og[(size_t)(rowb + mt * 16 + hi * 8 + r) * HIDDEN + chv + nt * 16 + n] = acc[r];
      }
    }

    // ---- S += K^T @ U'  (f32 VGPR accumulators, never rounded) ----
#pragma unroll
    for (int dvt = 0; dvt < 8; ++dvt) {
#pragma unroll
      for (int kk = 0; kk < 2; ++kk) {
        v16bf a = load_frag_a(sKt, CHK, wv * 16, kk * 32, lane);
        v16bf bb = load_frag_b(sWU, CHK, dvt * 16, kk * 32, lane);
        Sacc[dvt] = wmma_bf16(a, bb, Sacc[dvt]);
      }
    }
    __syncthreads();  // step-9 reads of sSt complete before refresh
    {
      int n = lane & 15, hi = lane >> 4;
#pragma unroll
      for (int dvt = 0; dvt < 8; ++dvt)
        store8_bf16(&sSt[(size_t)(dvt * 16 + n) * DKD + wv * 16 + hi * 8], Sacc[dvt]);
    }
    __syncthreads();
  }
}

// ---------------------------------------------------------------------------
// RMS norm over DV=256 per (row, head), bf16 out
// ---------------------------------------------------------------------------
__global__ __launch_bounds__(256) void rmsnorm_kernel(
    const float* __restrict__ o, const float* __restrict__ w,
    unsigned short* __restrict__ on) {
  int id = blockIdx.x * 8 + (threadIdx.x >> 5);
  int lane = threadIdx.x & 31;
  int row = id >> 2, h = id & 3;
  int c0 = h * DKD + lane * 8;
  const float* p = o + (size_t)row * HIDDEN + c0;
  f32x4 a = ((const f32x4*)p)[0];
  f32x4 bq = ((const f32x4*)p)[1];
  float xv[8] = {a.x, a.y, a.z, a.w, bq.x, bq.y, bq.z, bq.w};
  float ss = 0.f;
#pragma unroll
  for (int c = 0; c < 8; ++c) ss += xv[c] * xv[c];
#pragma unroll
  for (int m = 16; m; m >>= 1) ss += __shfl_xor(ss, m, 32);
  float sc = rsqrtf(ss * (1.0f / 256.0f) + 1e-5f);
  union { u32x4 q; unsigned short s8[8]; } u;
#pragma unroll
  for (int c = 0; c < 8; ++c) u.s8[c] = f2bf(xv[c] * sc * w[lane * 8 + c]);
  *(u32x4*)(on + (size_t)row * HIDDEN + c0) = u.q;
}

// ---------------------------------------------------------------------------
// Launch
// ---------------------------------------------------------------------------
extern "C" void kernel_launch(void* const* d_in, const int* in_sizes, int n_in,
                              void* d_out, int out_size, void* d_ws, size_t ws_size,
                              hipStream_t stream) {
  (void)in_sizes; (void)n_in; (void)out_size; (void)ws_size;
  const float* hs = (const float*)d_in[0];
  const float* Wq = (const float*)d_in[1];
  const float* Wk = (const float*)d_in[2];
  const float* Wv = (const float*)d_in[3];
  const float* Wb = (const float*)d_in[4];
  const float* cq = (const float*)d_in[5];
  const float* ck = (const float*)d_in[6];
  const float* cv = (const float*)d_in[7];
  const float* nw = (const float*)d_in[8];
  const float* Wo = (const float*)d_in[9];
  float* out = (float*)d_out;

  const size_t ROWS = 4096;  // B*S
  char* ws = (char*)d_ws;
  size_t off = 0;
  auto alloc = [&](size_t bytes) -> void* {
    void* p = ws + off;
    off = (off + bytes + 255) & ~(size_t)255;
    return p;
  };
  unsigned short* hsb = (unsigned short*)alloc(ROWS * HIDDEN * 2);
  unsigned short* WqT = (unsigned short*)alloc((size_t)HIDDEN * HIDDEN * 2);
  unsigned short* WkT = (unsigned short*)alloc((size_t)HIDDEN * HIDDEN * 2);
  unsigned short* WvT = (unsigned short*)alloc((size_t)HIDDEN * HIDDEN * 2);
  unsigned short* WoT = (unsigned short*)alloc((size_t)HIDDEN * HIDDEN * 2);
  float* qf = (float*)alloc(ROWS * HIDDEN * 4);
  float* kf = (float*)alloc(ROWS * HIDDEN * 4);
  float* vf = (float*)alloc(ROWS * HIDDEN * 4);
  unsigned short* qc = (unsigned short*)alloc(ROWS * HIDDEN * 2);
  unsigned short* kc = (unsigned short*)alloc(ROWS * HIDDEN * 2);
  unsigned short* vc = (unsigned short*)alloc(ROWS * HIDDEN * 2);
  float* beta = (float*)alloc(ROWS * NHEAD * 4);
  float* od = (float*)alloc(ROWS * HIDDEN * 4);
  unsigned short* on = (unsigned short*)alloc(ROWS * HIDDEN * 2);

  // 1) convert / transpose to bf16
  f32_to_bf16_kernel<<<4096, 256, 0, stream>>>(hs, hsb, (int)(ROWS * HIDDEN));
  transpose_to_bf16_kernel<<<4096, 256, 0, stream>>>(Wq, WqT, HIDDEN, HIDDEN);
  transpose_to_bf16_kernel<<<4096, 256, 0, stream>>>(Wk, WkT, HIDDEN, HIDDEN);
  transpose_to_bf16_kernel<<<4096, 256, 0, stream>>>(Wv, WvT, HIDDEN, HIDDEN);
  transpose_to_bf16_kernel<<<4096, 256, 0, stream>>>(Wo, WoT, HIDDEN, HIDDEN);

  // 2) q/k/v projections (WMMA bf16)
  dim3 gg(HIDDEN / 64, ROWS / 128);
  gemm_bf16<<<gg, 256, 0, stream>>>(hsb, WqT, qf, (int)ROWS, HIDDEN, HIDDEN);
  gemm_bf16<<<gg, 256, 0, stream>>>(hsb, WkT, kf, (int)ROWS, HIDDEN, HIDDEN);
  gemm_bf16<<<gg, 256, 0, stream>>>(hsb, WvT, vf, (int)ROWS, HIDDEN, HIDDEN);

  // 3) beta, conv+silu+l2norm
  beta_kernel<<<512, 256, 0, stream>>>(hs, Wb, beta);
  conv_silu_norm_kernel<<<6144, 256, 0, stream>>>(qf, kf, vf, cq, ck, cv, qc, kc, vc);

  // 4) chunked delta rule (16 WGs, 233KB LDS each)
  size_t ldsB = (size_t)(3 * 64 * DKD + DVSL * CHK + DVSL * DKD + CHK * CHK + 64 * DKD) * 2
              + (size_t)(64 * 66 + 64) * 4;
  delta_kernel<<<16, 512, ldsB, stream>>>(qc, kc, vc, beta, od);

  // 5) rms norm + output projection
  rmsnorm_kernel<<<2048, 256, 0, stream>>>(od, nw, on);
  gemm_bf16<<<gg, 256, 0, stream>>>(on, WoT, out, (int)ROWS, HIDDEN, HIDDEN);
}
